// RasterizedGaussianGoalMapWithUNetGridBCE_1821066134174
// MI455X (gfx1250) — matrix operand order repaired and checked by reading, not compiled, not run on hardware
//
#include <hip/hip_runtime.h>
#include <hip/hip_bf16.h>
#include <math.h>

// CDNA5 (gfx1250) wave32 WMMA types
typedef __attribute__((ext_vector_type(16))) _Float16 v16h;
typedef __attribute__((ext_vector_type(8)))  float    v8f;

#define HFULL 112
#define CROP  28
#define CW    56     // cropped window size
#define MOBJ  64     // objects per batch (K dimension)
#define PADW  64     // 56 padded to 64 for 16x16 tiling

// ---------------------------------------------------------------------------
// Kernel 0: zero the aux_loss accumulator in workspace
// ---------------------------------------------------------------------------
__global__ void init_ws_kernel(float* __restrict__ ws) { ws[0] = 0.0f; }

// ---------------------------------------------------------------------------
// Kernel 1: fused goal-map + object-Gaussian (WMMA rank-64 GEMM) + BCE sum.
// One block per batch element. 256 threads = 8 waves; 16 output tiles of
// 16x16 over the padded 64x64 crop window -> 2 tiles per wave, K=64 in two
// v_wmma_f32_16x16x32_f16 steps.
//
// Separable-Gaussian insight: exp(-(dr^2+dc^2)/8) = er(r)*ec(c), so
//   obj_sum[b,r,c] = sum_m valid_m * er[m,r] * ec[m,c]
// is a rank-64 GEMM: [56x64] @ [64x56] per batch -> WMMA.
//
// LDS layout matches CDNA5 16-bit WMMA VGPR layouts with contiguous dword
// (and b128-coalescible) loads:
//   SA[r][m] = valid_m * exp(-((r+28)-xo_m)^2 / 8)   (A matrix, MxK = 64x64)
//   SB[c][m] = exp(-((c+28)-yo_m)^2 / 8)             (B^T storage: K contiguous)
// ---------------------------------------------------------------------------
__global__ __launch_bounds__(256) void goalmap_bce_kernel(
    const int*   __restrict__ goal_pixel,   // [B,2]  (col,row)
    const int*   __restrict__ obj_list,     // [B,M,2]
    const int*   __restrict__ obj_num,      // [B]
    const int*   __restrict__ road_mask,    // [B,112,112]
    const float* __restrict__ goal_logits,  // [B,56,56]
    float*       __restrict__ aux_ws)       // [1] accumulator
{
    __shared__ __align__(16) _Float16 SA[PADW * MOBJ];
    __shared__ __align__(16) _Float16 SB[PADW * MOBJ];
    __shared__ float s_red;

    const int b   = blockIdx.x;
    const int tid = threadIdx.x;
    if (tid == 0) s_red = 0.0f;

    const int nvalid = obj_num[b];

    // Build separable exp tables (4096 entries each; 16 per thread)
    for (int idx = tid; idx < PADW * MOBJ; idx += 256) {
        const int rc = idx >> 6;    // row (SA) / col (SB), 0..63
        const int m  = idx & 63;    // object index (K)
        const float xo = (float)obj_list[(b * MOBJ + m) * 2 + 1]; // row coord
        const float yo = (float)obj_list[(b * MOBJ + m) * 2 + 0]; // col coord
        const float da = (float)(rc + CROP) - xo;
        const float db = (float)(rc + CROP) - yo;
        float ea = __expf(-da * da * 0.125f);   // sigma=2 -> 1/(2s^2)=0.125
        float eb = __expf(-db * db * 0.125f);
        if (m >= nvalid) ea = 0.0f;             // fold validity mask into A
        if (rc >= CW)   { ea = 0.0f; eb = 0.0f; } // zero padding rows/cols
        SA[idx] = (_Float16)ea;
        SB[idx] = (_Float16)eb;
    }
    __syncthreads();

    const float xL = (float)goal_pixel[b * 2 + 1]; // goal row
    const float yL = (float)goal_pixel[b * 2 + 0]; // goal col

    const int wave = tid >> 5;
    const int lane = tid & 31;
    const int half = lane >> 4;   // lane half selects K block / M block
    const int mrow = lane & 15;   // M (A) or N (B/C/D) index within tile

    union Frag { v16h h; unsigned int u[8]; };

    float acc = 0.0f;

#pragma unroll
    for (int tt = 0; tt < 2; ++tt) {
        const int t      = wave + tt * 8;      // tile 0..15
        const int tile_r = (t >> 2) * 16;
        const int tile_c = (t & 3) * 16;
        const int col    = tile_c + mrow;      // B/C/D column N = lane%16

        v8f c = {};
#pragma unroll
        for (int kk = 0; kk < 2; ++kk) {       // K = 64 in two 32-steps
            Frag a, bf;
            // A 16x32 f16 layout: lanes 0-15 (M=0..15): VGPR0-3 K=0..7,
            // VGPR4-7 K=16..23; lanes 16-31: +8. Dword j -> Kbase contiguous.
#pragma unroll
            for (int j = 0; j < 8; ++j) {
                const int kbase = 2 * j + ((j >= 4) ? 8 : 0) + 8 * half;
                a.u[j] = *(const unsigned int*)
                    &SA[(tile_r + mrow) * MOBJ + kk * 32 + kbase];
            }
            // B 32x16 f16 layout: lane = N, same K order as A (half adds 8).
#pragma unroll
            for (int j = 0; j < 8; ++j) {
                bf.u[j] = *(const unsigned int*)
                    &SB[col * MOBJ + kk * 32 + 2 * j + ((j >= 4) ? 8 : 0) + 8 * half];
            }
            c = __builtin_amdgcn_wmma_f32_16x16x32_f16(
                    /*neg_a=*/false, a.h, /*neg_b=*/false, bf.h,
                    /*c_mod=*/(short)0, c, /*reuse_a=*/false, /*reuse_b=*/false);
        }

        // Epilogue: C/D layout -> VGPR v holds M = v + 8*half, N = lane%16
#pragma unroll
        for (int v = 0; v < 8; ++v) {
            const int r = tile_r + v + 8 * half;
            if (r < CW && col < CW) {
                const float obj_sum = c[v];
                const float dr = (float)(r   + CROP) - xL;
                const float dc = (float)(col + CROP) - yL;
                const float g  = __expf(-(dr * dr + dc * dc) * 0.125f);
                float channel = 0.5f + 0.5f * g - 0.5f * obj_sum;
                if (road_mask[(b * HFULL + (r + CROP)) * HFULL + (col + CROP)] == 0)
                    channel = 0.0f;
                const float x = goal_logits[(b * CW + r) * CW + col];
                // logaddexp(0,x) = max(x,0) + log(1 + exp(-|x|))
                // hardware transcendentals (v_exp_f32 + v_log_f32) instead of
                // the ~40-instruction ocml log1pf expansion.
                const float lae = fmaxf(x, 0.0f) + __logf(1.0f + __expf(-fabsf(x)));
                acc += lae - x * channel;
            }
        }
    }

    atomicAdd(&s_red, acc);     // ds_add_f32
    __syncthreads();
    if (tid == 0) atomicAdd(aux_ws, s_red);  // one global atomic per block
}

// ---------------------------------------------------------------------------
// Kernel 2: MLP + MSE (motion_loss) and final output assembly.
// One block of 128 threads: thread t = batch row t.
// ---------------------------------------------------------------------------
__global__ __launch_bounds__(128) void motion_finalize_kernel(
    const int*   __restrict__ goal_pixel,  // [128,2]
    const float* __restrict__ tpos,        // [128,12,2]
    const float* __restrict__ tyaw,        // [128,12,1]
    const float* __restrict__ w1,          // [2,64]
    const float* __restrict__ b1,          // [64]
    const float* __restrict__ w2,          // [64,36]
    const float* __restrict__ b2,          // [36]
    const float* __restrict__ aux_ws,      // [1]
    float*       __restrict__ out)         // [3] = loss, aux, motion
{
    __shared__ float red[128];
    const int t = threadIdx.x;

    const float x0 = (float)goal_pixel[t * 2 + 0];
    const float x1 = (float)goal_pixel[t * 2 + 1];

    float hid[64];
#pragma unroll
    for (int j = 0; j < 64; ++j) {
        const float h = fmaf(x0, w1[j], fmaf(x1, w1[64 + j], b1[j]));
        hid[j] = fmaxf(h, 0.0f);
    }

    float ss = 0.0f;
#pragma unroll 3
    for (int ts = 0; ts < 12; ++ts) {       // time steps, 3 comps each
#pragma unroll
        for (int comp = 0; comp < 3; ++comp) {
            const int k = ts * 3 + comp;
            float o = b2[k];
#pragma unroll
            for (int j = 0; j < 64; ++j) o = fmaf(hid[j], w2[j * 36 + k], o);
            const float tgt = (comp < 2) ? tpos[(t * 12 + ts) * 2 + comp]
                                         : tyaw[t * 12 + ts];
            const float d = o - tgt;
            ss = fmaf(d, d, ss);
        }
    }

    red[t] = ss;
    __syncthreads();
#pragma unroll
    for (int s = 64; s > 0; s >>= 1) {
        if (t < s) red[t] += red[t + s];
        __syncthreads();
    }
    if (t == 0) {
        const float motion = red[0] * (1.0f / (128.0f * 36.0f));
        const float aux    = aux_ws[0];
        out[0] = motion + aux;
        out[1] = aux;
        out[2] = motion;
    }
}

// ---------------------------------------------------------------------------
extern "C" void kernel_launch(void* const* d_in, const int* in_sizes, int n_in,
                              void* d_out, int out_size, void* d_ws, size_t ws_size,
                              hipStream_t stream) {
    const int*   goal_pixel  = (const int*)  d_in[0];
    const int*   obj_list    = (const int*)  d_in[1];
    const int*   obj_num     = (const int*)  d_in[2];
    const int*   road_mask   = (const int*)  d_in[3];
    const float* goal_logits = (const float*)d_in[4];
    const float* tpos        = (const float*)d_in[5];
    const float* tyaw        = (const float*)d_in[6];
    const float* w1          = (const float*)d_in[7];
    const float* b1          = (const float*)d_in[8];
    const float* w2          = (const float*)d_in[9];
    const float* b2          = (const float*)d_in[10];
    float* out = (float*)d_out;
    float* ws  = (float*)d_ws;

    hipLaunchKernelGGL(init_ws_kernel, dim3(1), dim3(1), 0, stream, ws);
    hipLaunchKernelGGL(goalmap_bce_kernel, dim3(128), dim3(256), 0, stream,
                       goal_pixel, obj_list, obj_num, road_mask, goal_logits, ws);
    hipLaunchKernelGGL(motion_finalize_kernel, dim3(1), dim3(128), 0, stream,
                       goal_pixel, tpos, tyaw, w1, b1, w2, b2, ws, out);
}
